// LSTM_56530359550084
// MI455X (gfx1250) — matrix operand order repaired and checked by reading, not compiled, not run on hardware
//
#include <hip/hip_runtime.h>
#include <math.h>

// ---------------------------------------------------------------------------
// LSTM (B=64, T=256, I=512, H=1024) for gfx1250 (MI455X), wave32 + WMMA bf16.
//
//  * Pre-pass: x -> bf16 [T][B][I]; W_x/W_h -> bf16 packed per-block weight
//    slabs in WMMA B-fragment layout: slab[blk] = [Wh 128KB | Wx 64KB], so the
//    recurrent kernel can TDM-copy one contiguous 192KB tile into LDS once.
//  * Persistent recurrent kernel: 64 blocks, each owns 16 h-columns x 4 gates.
//    Weights live in LDS for all 256 steps (ds_load_b128 B fragments); h_prev
//    rows stream from L2 (global_load_b128 A fragments); fp32 WMMA accum;
//    LDS gate-tile exchange; cell update with c in registers; ping-pong bf16
//    h buffer; device-wide barrier via agent-scope atomic counter.
// ---------------------------------------------------------------------------

typedef __attribute__((ext_vector_type(16))) __bf16 v16bf;
typedef __attribute__((ext_vector_type(8)))  float  v8f;
typedef __attribute__((ext_vector_type(4)))  unsigned u32x4;
typedef __attribute__((ext_vector_type(4)))  unsigned uint32x4_t;
typedef __attribute__((ext_vector_type(8)))  int      int32x8_t;
typedef __attribute__((ext_vector_type(4)))  int      int32x4_t;

#define B_    64
#define T_    256
#define I_    512
#define H_    1024
#define N_    4096      // 4 gates * H
#define NBLK  64
#define NTH   256

#define WH_SLAB_BYTES 131072u            // 32 kc * 4 gates * 32 lanes * 32 B
#define WX_SLAB_BYTES 65536u             // 16 kc * 4 gates * 32 lanes * 32 B
#define WSLAB_BYTES   (WH_SLAB_BYTES + WX_SLAB_BYTES)      // 192 KB per block
#define GALL_BYTES    (4 * B_ * 16 * 4)                    // 16 KB
#define SMEM_BYTES    (WSLAB_BYTES + GALL_BYTES)           // 208 KB dynamic LDS

union AFrag { u32x4 q[2]; v16bf v; };
union BFrag { u32x4 q[2]; v16bf v; };
union Pair  { __bf16 h[2]; unsigned u; };

__device__ __forceinline__ float sigmoid_f(float x) {
    return 1.0f / (1.0f + __expf(-x));
}

// ---------------------------------------------------------------------------
// Pack x[b][t][i] (f32) -> xT[t][b][i] (bf16)
// ---------------------------------------------------------------------------
__global__ __launch_bounds__(256) void pack_x_kernel(const float* __restrict__ x,
                                                     __bf16* __restrict__ xT) {
    long idx = (long)blockIdx.x * blockDim.x + threadIdx.x;   // B*T*I threads
    int i = (int)(idx % I_);
    int t = (int)((idx / I_) % T_);
    int b = (int)(idx / ((long)I_ * T_));
    xT[((long)t * B_ + b) * I_ + i] = (__bf16)x[((long)b * T_ + t) * I_ + i];
}

// ---------------------------------------------------------------------------
// Pack W_h[g][k_out][k_sum] into per-block slabs (WMMA B-fragment layout).
// Slab dword layout (within block):  [kc:32][g:4][lane:32][p:8]
//   k_out = blk*16 + (lane&15)
//   k_sum = kc*32 + (lane&16 ? 16 : 0) + 2*p  (+0,+1 bf16 pair)
// ---------------------------------------------------------------------------
__global__ __launch_bounds__(256) void pack_wh_kernel(const float* __restrict__ Wh,
                                                      unsigned* __restrict__ Wp) {
    long d = (long)blockIdx.x * blockDim.x + threadIdx.x;     // 64 * 32768 threads
    int p    = (int)(d & 7);
    int lane = (int)((d >> 3) & 31);
    int g    = (int)((d >> 8) & 3);
    int kc   = (int)((d >> 10) & 31);
    int blk  = (int)(d >> 15);
    int kout = blk * 16 + (lane & 15);
    int ks   = kc * 32 + ((lane & 16) ? 16 : 0) + 2 * p;
    const float* src = Wh + ((long)(g * H_ + kout)) * H_ + ks;
    Pair pr; pr.h[0] = (__bf16)src[0]; pr.h[1] = (__bf16)src[1];
    Wp[(size_t)blk * (WSLAB_BYTES / 4) + (d & 32767)] = pr.u;
}

// Same for W_x (K = 512): slab dwords [kc:16][g:4][lane:32][p:8], after Wh part.
__global__ __launch_bounds__(256) void pack_wx_kernel(const float* __restrict__ Wx,
                                                      unsigned* __restrict__ Wp) {
    long d = (long)blockIdx.x * blockDim.x + threadIdx.x;     // 64 * 16384 threads
    int p    = (int)(d & 7);
    int lane = (int)((d >> 3) & 31);
    int g    = (int)((d >> 8) & 3);
    int kc   = (int)((d >> 10) & 15);
    int blk  = (int)(d >> 14);
    int kout = blk * 16 + (lane & 15);
    int ks   = kc * 32 + ((lane & 16) ? 16 : 0) + 2 * p;
    const float* src = Wx + ((long)(g * H_ + kout)) * I_ + ks;
    Pair pr; pr.h[0] = (__bf16)src[0]; pr.h[1] = (__bf16)src[1];
    Wp[(size_t)blk * (WSLAB_BYTES / 4) + (WH_SLAB_BYTES / 4) + (d & 16383)] = pr.u;
}

// ---------------------------------------------------------------------------
// bias vector + zero h ping-pong buffers + zero barrier counter.
// ---------------------------------------------------------------------------
__global__ __launch_bounds__(256) void init_kernel(const float* __restrict__ b_x,
                                                   const float* __restrict__ b_h,
                                                   const float* __restrict__ w_B,
                                                   const float* __restrict__ b_B,
                                                   float* __restrict__ biasv,
                                                   __bf16* __restrict__ hbuf,
                                                   unsigned* __restrict__ counter) {
    int tid = blockIdx.x * blockDim.x + threadIdx.x;
    if (tid < N_) biasv[tid] = b_x[tid] + b_h[tid] + w_B[tid] + b_B[tid];
    if (tid < 2 * B_ * H_) hbuf[tid] = (__bf16)0.0f;
    if (tid == 0) *counter = 0u;
}

// ---------------------------------------------------------------------------
// Persistent recurrent kernel. Dynamic LDS: [weight slab 192KB][gAll 16KB].
// ---------------------------------------------------------------------------
__global__ __launch_bounds__(NTH, 1)
void lstm_recur_kernel(const __bf16* __restrict__ xT,
                       const unsigned* __restrict__ Wp,
                       const float* __restrict__ biasv,
                       __bf16* __restrict__ hbuf,
                       float* __restrict__ out,
                       unsigned* __restrict__ counter) {
    extern __shared__ char smem[];
    float* gA = (float*)(smem + WSLAB_BYTES);   // [4][64][16]

    const int tid   = threadIdx.x;
    const int lane  = tid & 31;
    const int wave  = tid >> 5;           // 0..7
    const int blk   = blockIdx.x;         // 0..63 -> h columns [blk*16, +16)
    const int mt    = wave & 3;           // batch tile (16 rows each)
    const int gp    = wave >> 2;          // gate pair: {0,1} or {2,3}
    const int g0    = 2 * gp, g1 = 2 * gp + 1;
    const int hbase = blk * 16;

    const int mrow = mt * 16 + (lane & 15);
    const int ahi  = (lane & 16) ? 16 : 0;

    // ---- stage this block's 192KB weight slab into LDS (once) -----------
    const char* wsrc = (const char*)Wp + (size_t)blk * WSLAB_BYTES;
#if __has_builtin(__builtin_amdgcn_tensor_load_to_lds) && \
    __has_builtin(__builtin_amdgcn_s_wait_tensorcnt)
    if (tid < 32) {
        // D# group0: count=1 valid, lds_addr=0 (dynamic LDS base, no statics),
        // 57-bit global byte address, type=2.
        unsigned long long ga = (unsigned long long)(size_t)wsrc;
        uint32x4_t g0d;
        g0d.x = 1u;
        g0d.y = 0u;
        g0d.z = (unsigned)(ga & 0xFFFFFFFFu);
        g0d.w = (unsigned)((ga >> 32) & 0x01FFFFFFu) | (2u << 30);
        // D# group1: data_size=8B, 2D tensor/tile 4096 x 6 (= 192KB),
        // dim0 stride 4096 elements, no multicast/pad/iterate/barrier.
        int32x8_t g1d;
        g1d[0] = 0x30000;      // data_size=3 (8B)
        g1d[1] = 0x10000000;   // tensor_dim0[15:0]=4096 << 16
        g1d[2] = 0x60000;      // tensor_dim1[15:0]=6 << 16
        g1d[3] = 0x10000000;   // tile_dim0=4096 << 16
        g1d[4] = 6;            // tile_dim1=6
        g1d[5] = 4096;         // tensor_dim0_stride low32
        g1d[6] = 0;
        g1d[7] = 0;
        int32x4_t z4 = {0, 0, 0, 0};
        int32x8_t z8 = {0, 0, 0, 0, 0, 0, 0, 0};
        // 6-arg form (clang-23 / therock headers): groups 0..3 + pad + cpol.
        __builtin_amdgcn_tensor_load_to_lds(g0d, g1d, z4, z4, z8, 0);
        __builtin_amdgcn_s_wait_tensorcnt(0);
    }
#else
    {
        u32x4* dst = (u32x4*)smem;
        const u32x4* src = (const u32x4*)wsrc;
        for (int i = tid; i < (int)(WSLAB_BYTES / 16); i += NTH) dst[i] = src[i];
    }
#endif
    __syncthreads();

    // Per-wave LDS base for B fragments (gate pair), byte offsets.
    const char* bh_base = smem + (size_t)(g0 * 1024 + lane * 32);

    // Cell state in registers: thread owns 4 consecutive (m, hc) elements.
    const int em  = tid >> 2;             // batch row 0..63
    const int ehc = (tid & 3) * 4;        // hcol base 0..12
    float c_reg[4] = {0.f, 0.f, 0.f, 0.f};

    for (int t = 0; t < T_; ++t) {
        const __bf16* hprev = hbuf + (size_t)(t & 1) * (B_ * H_);
        __bf16*       hnext = hbuf + (size_t)((t + 1) & 1) * (B_ * H_);

        if (t + 1 < T_)   // warm L2/L0 for next step's x rows
            __builtin_prefetch(xT + ((size_t)(t + 1) * B_ + mrow) * I_, 0, 1);

        v8f acc0 = {}; v8f acc1 = {};

        // ---- recurrent part: K = 1024, 32 chunks of 32 ------------------
        {
            const __bf16* ap = hprev + (size_t)mrow * H_ + ahi;
            const char*   bp = bh_base;
            #pragma unroll 2
            for (int kc = 0; kc < H_ / 32; ++kc) {
                AFrag a;
                a.q[0] = *(const u32x4*)(ap);
                a.q[1] = *(const u32x4*)(ap + 8);
                BFrag b0, b1;
                b0.q[0] = *(const u32x4*)(bp);
                b0.q[1] = *(const u32x4*)(bp + 16);
                b1.q[0] = *(const u32x4*)(bp + 1024);
                b1.q[1] = *(const u32x4*)(bp + 1040);
                acc0 = __builtin_amdgcn_wmma_f32_16x16x32_bf16(
                    false, a.v, false, b0.v, (short)0, acc0, false, false);
                acc1 = __builtin_amdgcn_wmma_f32_16x16x32_bf16(
                    false, a.v, false, b1.v, (short)0, acc1, false, false);
                ap += 32;
                bp += 4096;
            }
        }

        // ---- input part: K = 512, 16 chunks of 32 -----------------------
        {
            const __bf16* xp = xT + ((size_t)t * B_ + mrow) * I_ + ahi;
            const char*   bp = bh_base + WH_SLAB_BYTES;
            #pragma unroll 2
            for (int kc = 0; kc < I_ / 32; ++kc) {
                AFrag a;
                a.q[0] = *(const u32x4*)(xp);
                a.q[1] = *(const u32x4*)(xp + 8);
                BFrag b0, b1;
                b0.q[0] = *(const u32x4*)(bp);
                b0.q[1] = *(const u32x4*)(bp + 16);
                b1.q[0] = *(const u32x4*)(bp + 1024);
                b1.q[1] = *(const u32x4*)(bp + 1040);
                acc0 = __builtin_amdgcn_wmma_f32_16x16x32_bf16(
                    false, a.v, false, b0.v, (short)0, acc0, false, false);
                acc1 = __builtin_amdgcn_wmma_f32_16x16x32_bf16(
                    false, a.v, false, b1.v, (short)0, acc1, false, false);
                xp += 32;
                bp += 4096;
            }
        }

        // ---- dump accumulator tiles to LDS ------------------------------
        // C/D layout: VGPR r, lane l -> M = r + 8*(l>=16), N = l&15
        {
            const int nc = lane & 15;
            const int mb = mt * 16 + ((lane & 16) ? 8 : 0);
            #pragma unroll
            for (int r = 0; r < 8; ++r) {
                gA[((g0 * B_) + mb + r) * 16 + nc] = acc0[r];
                gA[((g1 * B_) + mb + r) * 16 + nc] = acc1[r];
            }
        }
        __syncthreads();

        // ---- cell update (f,i,g,o = gates 0..3) -------------------------
        #pragma unroll
        for (int j = 0; j < 4; ++j) {
            const int hc = ehc + j;
            const int nglob = hbase + hc;
            float pf = gA[(0 * B_ + em) * 16 + hc] + biasv[0 * H_ + nglob];
            float pi = gA[(1 * B_ + em) * 16 + hc] + biasv[1 * H_ + nglob];
            float pg = gA[(2 * B_ + em) * 16 + hc] + biasv[2 * H_ + nglob];
            float po = gA[(3 * B_ + em) * 16 + hc] + biasv[3 * H_ + nglob];
            float f = sigmoid_f(pf);
            float i = sigmoid_f(pi);
            float g = tanhf(pg);
            float o = sigmoid_f(po);
            float cn = f * c_reg[j] + i * g;
            c_reg[j] = cn;
            float hn = o * tanhf(cn);
            out[((size_t)em * T_ + t) * H_ + nglob] = hn;
            hnext[(size_t)em * H_ + nglob] = (__bf16)hn;
        }

        // ---- device-wide barrier (monotonic counter) --------------------
        __threadfence();
        __syncthreads();
        if (tid == 0) {
            __hip_atomic_fetch_add(counter, 1u, __ATOMIC_ACQ_REL,
                                   __HIP_MEMORY_SCOPE_AGENT);
            const unsigned target = (unsigned)(NBLK * (t + 1));
            while (__hip_atomic_load(counter, __ATOMIC_ACQUIRE,
                                     __HIP_MEMORY_SCOPE_AGENT) < target) {
                __builtin_amdgcn_s_sleep(2);
            }
        }
        __syncthreads();
    }
}

// ---------------------------------------------------------------------------
// Host-side launcher
// ---------------------------------------------------------------------------
extern "C" void kernel_launch(void* const* d_in, const int* in_sizes, int n_in,
                              void* d_out, int out_size, void* d_ws, size_t ws_size,
                              hipStream_t stream) {
    const float* x   = (const float*)d_in[0];   // [B,T,I]
    const float* W_x = (const float*)d_in[1];   // [4,H,I]
    const float* b_x = (const float*)d_in[2];   // [4,H]
    const float* W_h = (const float*)d_in[3];   // [4,H,H]
    const float* b_h = (const float*)d_in[4];   // [4,H]
    const float* w_B = (const float*)d_in[5];   // [4,H]
    const float* b_B = (const float*)d_in[6];   // [4,H]
    float* out = (float*)d_out;                 // [B,T,H]

    char* ws = (char*)d_ws;
    size_t off = 0;
    __bf16*   xT      = (__bf16*)(ws + off);   off += (size_t)T_ * B_ * I_ * 2;   // 16 MB
    unsigned* Wp      = (unsigned*)(ws + off); off += (size_t)NBLK * WSLAB_BYTES; // 12 MB
    float*    biasv   = (float*)(ws + off);    off += (size_t)N_ * 4;
    __bf16*   hbuf    = (__bf16*)(ws + off);   off += (size_t)2 * B_ * H_ * 2;
    unsigned* counter = (unsigned*)(ws + off); off += 256;
    (void)ws_size; (void)in_sizes; (void)n_in; (void)out_size;

    pack_x_kernel <<<(B_ * T_ * I_) / 256,            256, 0, stream>>>(x, xT);
    pack_wh_kernel<<<(NBLK * 32768) / 256,            256, 0, stream>>>(W_h, Wp);
    pack_wx_kernel<<<(NBLK * 16384) / 256,            256, 0, stream>>>(W_x, Wp);
    init_kernel   <<<(2 * B_ * H_) / 256,             256, 0, stream>>>(b_x, b_h, w_B, b_B,
                                                                        biasv, hbuf, counter);
    lstm_recur_kernel<<<NBLK, NTH, SMEM_BYTES, stream>>>(xT, Wp, biasv, hbuf, out, counter);
}
